// QWenBlock_59219009077593
// MI455X (gfx1250) — compile-verified, matrix operand order checked by reading, not run on hardware
//
#include <hip/hip_runtime.h>
#include <stdint.h>

#define B_   2
#define S_   2048
#define D_   4096
#define H_   32
#define HD_  128
#define FF_  11008
#define TOK_ (B_ * S_)
#define QKVN_ (3 * D_)

typedef __attribute__((ext_vector_type(16))) __bf16 v16bf;
typedef __attribute__((ext_vector_type(8)))  float  v8f;

union Frag { v16bf v; uint4 q[2]; unsigned int u[8]; };

__device__ inline unsigned short f2bf(float f) {
  union { float f; unsigned int u; } c; c.f = f;
  unsigned int r = c.u + 0x7FFFu + ((c.u >> 16) & 1u);   // round-to-nearest-even
  return (unsigned short)(r >> 16);
}
__device__ inline unsigned int pack2(float a, float b) {
  return (unsigned int)f2bf(a) | ((unsigned int)f2bf(b) << 16);
}
__device__ inline v8f wmma_bf16(const Frag& a, const Frag& b, v8f c) {
  return __builtin_amdgcn_wmma_f32_16x16x32_bf16(false, a.v, false, b.v,
                                                 (short)0, c, false, false);
}

// gfx1250 async global->LDS copy (16B per lane), tracked by ASYNCcnt.
__device__ inline void async_g2l_b128(unsigned lds_addr, const void* g) {
  asm volatile("global_load_async_to_lds_b128 %0, %1, off"
               :: "v"(lds_addr), "v"((unsigned long long)(uintptr_t)g)
               : "memory");
}
__device__ inline void wait_async0() {
  asm volatile("s_wait_asynccnt 0x0" ::: "memory");
}
__device__ inline unsigned lds_addr_of(const void* p) {
  return (unsigned)(uintptr_t)p;   // LDS aperture: low 32 bits = LDS byte address
}

// ------------------------------------------------ fp32 [K,N] -> bf16 transposed [N,K]
__global__ __launch_bounds__(256) void cvt_transpose(const float* __restrict__ src,
                                                     unsigned short* __restrict__ dst,
                                                     int K, int N) {
  __shared__ unsigned short t[32][40];
  const int n0 = blockIdx.x * 32, k0 = blockIdx.y * 32;
  const int tx = threadIdx.x & 31, ty = threadIdx.x >> 5;
#pragma unroll
  for (int i = 0; i < 32; i += 8)
    t[ty + i][tx] = f2bf(src[(size_t)(k0 + ty + i) * N + n0 + tx]);
  __syncthreads();
#pragma unroll
  for (int i = 0; i < 32; i += 8)
    dst[(size_t)(n0 + ty + i) * K + k0 + tx] = t[tx][ty + i];
}

// ------------------------------------------------ RMSNorm -> bf16
__global__ __launch_bounds__(256) void rmsnorm_kernel(const float* __restrict__ x,
                                                      const float* __restrict__ w,
                                                      unsigned short* __restrict__ out) {
  __shared__ float red[256];
  const int t = threadIdx.x;
  const float* xr = x + (size_t)blockIdx.x * D_;
  float s = 0.f;
  for (int i = t; i < D_; i += 256) { float v = xr[i]; s += v * v; }
  red[t] = s; __syncthreads();
  for (int o = 128; o > 0; o >>= 1) {
    if (t < o) red[t] += red[t + o];
    __syncthreads();
  }
  float rr = rsqrtf(red[0] * (1.0f / D_) + 1e-6f);
  unsigned short* orow = out + (size_t)blockIdx.x * D_;
  for (int i = t; i < D_; i += 256) orow[i] = f2bf(xr[i] * rr * w[i]);
}

// ------------------------------------------------ RoPE (in place on q,k)
__global__ void rope_kernel(const int* __restrict__ pos, float* __restrict__ qkv) {
  int idx = blockIdx.x * blockDim.x + threadIdx.x;
  if (idx >= TOK_ * H_ * (HD_ / 2)) return;
  int p = idx & 63;
  int h = (idx >> 6) & (H_ - 1);
  int t = idx >> 11;
  float fr = (float)pos[t] * exp2f(-(float)(2 * p) * (13.287712379549449f / 128.0f));
  float c = cosf(fr), sn = sinf(fr);
  float* qp = qkv + (size_t)t * QKVN_ + h * HD_;
  float* kp = qp + D_;
  float x1 = qp[p], x2 = qp[p + 64];
  qp[p] = x1 * c - x2 * sn; qp[p + 64] = x2 * c + x1 * sn;
  x1 = kp[p]; x2 = kp[p + 64];
  kp[p] = x1 * c - x2 * sn; kp[p + 64] = x2 * c + x1 * sn;
}

// ------------------------------------------------ K heads -> bf16 head-major [bh,s,hd]
__global__ void cvt_khead(const float* __restrict__ qkv, unsigned short* __restrict__ kbf) {
  size_t idx = ((size_t)blockIdx.x * blockDim.x + threadIdx.x) * 4;
  if (idx >= (size_t)TOK_ * D_) return;
  size_t d = idx & (HD_ - 1);
  size_t s = (idx / HD_) % S_;
  size_t bh = idx / ((size_t)S_ * HD_);
  size_t b = bh / H_, h = bh % H_;
  const float* src = qkv + (b * S_ + s) * QKVN_ + D_ + h * HD_ + d;
  float4 f = *(const float4*)src;
  uint2 o; o.x = pack2(f.x, f.y); o.y = pack2(f.z, f.w);
  *(uint2*)(kbf + idx) = o;
}

// ------------------------------------------------ V heads -> bf16 transposed [bh,hd,s]
__global__ __launch_bounds__(256) void vtrans(const float* __restrict__ qkv,
                                              unsigned short* __restrict__ vbt) {
  __shared__ unsigned short t[32][40];
  const int s0 = blockIdx.x * 32, d0 = blockIdx.y * 32;
  const int bh = blockIdx.z, b = bh / H_, h = bh % H_;
  const int tx = threadIdx.x & 31, ty = threadIdx.x >> 5;
#pragma unroll
  for (int i = 0; i < 32; i += 8)
    t[ty + i][tx] =
        f2bf(qkv[(size_t)(b * S_ + s0 + ty + i) * QKVN_ + 2 * D_ + h * HD_ + d0 + tx]);
  __syncthreads();
#pragma unroll
  for (int i = 0; i < 32; i += 8)
    vbt[((size_t)bh * HD_ + d0 + ty + i) * S_ + s0 + tx] = t[tx][ty + i];
}

// ------------------------------------------------ GEMM (bf16 WMMA, async double-buffered)
// B is pre-transposed [N][K].  MODE 0: +bias[col]   MODE 1: +res[row,col]
template <int MODE>
__global__ __launch_bounds__(256) void gemm_bf16(const unsigned short* __restrict__ A,
                                                 const unsigned short* __restrict__ Bt,
                                                 const float* __restrict__ aux,
                                                 float* __restrict__ C,
                                                 int M, int N, int K) {
  __shared__ unsigned short lds_a[2][64 * 40];    // [row][k], pad 8
  __shared__ unsigned short lds_b[2][128 * 40];   // [n][k],  pad 8
  const int tid = threadIdx.x;
  const int lane = tid & 31, w = tid >> 5;
  const int hi = lane >> 4, ln = lane & 15;
  const int waveM = w >> 2, waveN = w & 3;
  const int m0 = blockIdx.y * 64, n0 = blockIdx.x * 128;
  const int ar = tid >> 2, ac = (tid & 3) * 8;    // A: one 16B chunk / thread
  const int br = tid >> 1, bc = (tid & 1) * 16;   // B: two 16B chunks / thread
  const unsigned la0 = lds_addr_of(&lds_a[0][0]), la1 = lds_addr_of(&lds_a[1][0]);
  const unsigned lb0 = lds_addr_of(&lds_b[0][0]), lb1 = lds_addr_of(&lds_b[1][0]);

  auto stage = [&](int buf, int k0) {
    unsigned la = (buf ? la1 : la0) + (unsigned)(ar * 40 + ac) * 2;
    async_g2l_b128(la, A + (size_t)(m0 + ar) * K + k0 + ac);
    unsigned lb = (buf ? lb1 : lb0) + (unsigned)(br * 40 + bc) * 2;
    const unsigned short* gb = Bt + (size_t)(n0 + br) * K + k0 + bc;
    async_g2l_b128(lb, gb);
    async_g2l_b128(lb + 16, gb + 8);
  };

  v8f acc[2][2] = {};
  stage(0, 0);
  wait_async0();
  __syncthreads();
  int cur = 0;
  for (int k0 = 0; k0 < K; k0 += 32, cur ^= 1) {
    if (k0 + 32 < K) stage(cur ^ 1, k0 + 32);   // overlaps with WMMA below

    Frag a[2], b[2];
#pragma unroll
    for (int mi = 0; mi < 2; ++mi) {
      int base = (waveM * 32 + mi * 16 + ln) * 40 + hi * 8;
      a[mi].q[0] = *(const uint4*)(&lds_a[cur][base]);
      a[mi].q[1] = *(const uint4*)(&lds_a[cur][base + 16]);
    }
#pragma unroll
    for (int ni = 0; ni < 2; ++ni) {
      int base = (waveN * 32 + ni * 16 + ln) * 40 + hi * 8;
      b[ni].q[0] = *(const uint4*)(&lds_b[cur][base]);
      b[ni].q[1] = *(const uint4*)(&lds_b[cur][base + 16]);
    }
#pragma unroll
    for (int mi = 0; mi < 2; ++mi)
#pragma unroll
      for (int ni = 0; ni < 2; ++ni)
        acc[mi][ni] = wmma_bf16(a[mi], b[ni], acc[mi][ni]);

    wait_async0();
    __syncthreads();
  }
#pragma unroll
  for (int mi = 0; mi < 2; ++mi)
#pragma unroll
    for (int ni = 0; ni < 2; ++ni)
#pragma unroll
      for (int r = 0; r < 8; ++r) {
        int row = m0 + waveM * 32 + mi * 16 + r + hi * 8;
        int col = n0 + waveN * 32 + ni * 16 + ln;
        float v = acc[mi][ni][r];
        if (MODE == 0) v += aux[col];
        if (MODE == 1) v += aux[(size_t)row * N + col];
        C[(size_t)row * N + col] = v;
      }
}

// ------------------------------------------------ dual GEMM + SwiGLU -> bf16
__global__ __launch_bounds__(256) void gemm_swiglu(const unsigned short* __restrict__ A,
                                                   const unsigned short* __restrict__ B1t,
                                                   const unsigned short* __restrict__ B2t,
                                                   unsigned short* __restrict__ Out,
                                                   int M, int N, int K) {
  __shared__ unsigned short lds_a[2][64 * 40];
  __shared__ unsigned short lds_b1[2][128 * 40];
  __shared__ unsigned short lds_b2[2][128 * 40];
  const int tid = threadIdx.x;
  const int lane = tid & 31, w = tid >> 5;
  const int hi = lane >> 4, ln = lane & 15;
  const int waveM = w >> 2, waveN = w & 3;
  const int m0 = blockIdx.y * 64, n0 = blockIdx.x * 128;
  const int ar = tid >> 2, ac = (tid & 3) * 8;
  const int br = tid >> 1, bc = (tid & 1) * 16;
  const unsigned la0 = lds_addr_of(&lds_a[0][0]),  la1 = lds_addr_of(&lds_a[1][0]);
  const unsigned lb10 = lds_addr_of(&lds_b1[0][0]), lb11 = lds_addr_of(&lds_b1[1][0]);
  const unsigned lb20 = lds_addr_of(&lds_b2[0][0]), lb21 = lds_addr_of(&lds_b2[1][0]);

  auto stage = [&](int buf, int k0) {
    unsigned la = (buf ? la1 : la0) + (unsigned)(ar * 40 + ac) * 2;
    async_g2l_b128(la, A + (size_t)(m0 + ar) * K + k0 + ac);
    unsigned off = (unsigned)(br * 40 + bc) * 2;
    const unsigned short* g1 = B1t + (size_t)(n0 + br) * K + k0 + bc;
    const unsigned short* g2 = B2t + (size_t)(n0 + br) * K + k0 + bc;
    unsigned l1 = (buf ? lb11 : lb10) + off;
    unsigned l2 = (buf ? lb21 : lb20) + off;
    async_g2l_b128(l1, g1); async_g2l_b128(l1 + 16, g1 + 8);
    async_g2l_b128(l2, g2); async_g2l_b128(l2 + 16, g2 + 8);
  };

  v8f acc1[2][2] = {}, acc2[2][2] = {};
  stage(0, 0);
  wait_async0();
  __syncthreads();
  int cur = 0;
  for (int k0 = 0; k0 < K; k0 += 32, cur ^= 1) {
    if (k0 + 32 < K) stage(cur ^ 1, k0 + 32);

    Frag a[2], b1[2], b2[2];
#pragma unroll
    for (int mi = 0; mi < 2; ++mi) {
      int base = (waveM * 32 + mi * 16 + ln) * 40 + hi * 8;
      a[mi].q[0] = *(const uint4*)(&lds_a[cur][base]);
      a[mi].q[1] = *(const uint4*)(&lds_a[cur][base + 16]);
    }
#pragma unroll
    for (int ni = 0; ni < 2; ++ni) {
      int base = (waveN * 32 + ni * 16 + ln) * 40 + hi * 8;
      b1[ni].q[0] = *(const uint4*)(&lds_b1[cur][base]);
      b1[ni].q[1] = *(const uint4*)(&lds_b1[cur][base + 16]);
      b2[ni].q[0] = *(const uint4*)(&lds_b2[cur][base]);
      b2[ni].q[1] = *(const uint4*)(&lds_b2[cur][base + 16]);
    }
#pragma unroll
    for (int mi = 0; mi < 2; ++mi)
#pragma unroll
      for (int ni = 0; ni < 2; ++ni) {
        acc1[mi][ni] = wmma_bf16(a[mi], b1[ni], acc1[mi][ni]);
        acc2[mi][ni] = wmma_bf16(a[mi], b2[ni], acc2[mi][ni]);
      }
    wait_async0();
    __syncthreads();
  }
#pragma unroll
  for (int mi = 0; mi < 2; ++mi)
#pragma unroll
    for (int ni = 0; ni < 2; ++ni)
#pragma unroll
      for (int r = 0; r < 8; ++r) {
        int row = m0 + waveM * 32 + mi * 16 + r + hi * 8;
        int col = n0 + waveN * 32 + ni * 16 + ln;
        float x1 = acc1[mi][ni][r];
        float x2 = acc2[mi][ni][r];
        float sg = x2 * __builtin_amdgcn_rcpf(1.0f + __expf(-x2));  // fast silu
        Out[(size_t)row * N + col] = f2bf(x1 * sg);
      }
}

// ------------------------------------------------ flash attention -> bf16
__global__ __launch_bounds__(128) void attn_kernel(const float* __restrict__ qkv,
                                                   const unsigned short* __restrict__ kbf,
                                                   const unsigned short* __restrict__ vbt,
                                                   unsigned short* __restrict__ attn) {
  __shared__ unsigned short lds_k[64 * 136];    // [key][hd], pad 8
  __shared__ unsigned short lds_v[128 * 72];    // [hd][key], pad 8
  __shared__ unsigned short lds_p[4 * 16 * 72]; // per-wave P tiles [row][key]
  const int tid = threadIdx.x;
  const int lane = tid & 31, w = tid >> 5;
  const int hi = lane >> 4, ln = lane & 15;
  int bx = blockIdx.x;
  const int qt = bx % (S_ / 64); bx /= (S_ / 64);
  const int h = bx % H_; const int b = bx / H_;
  const int bh = b * H_ + h;
  const int qb = qt * 64;
  const unsigned lk0 = lds_addr_of(lds_k), lv0 = lds_addr_of(lds_v);

  // Q fragments (scale 1/sqrt(HD) folded in), kept in regs for whole kernel
  Frag qf[4];
  {
    const int qrow = qb + w * 16 + ln;
    const float* qp = qkv + (size_t)(b * S_ + qrow) * QKVN_ + h * HD_;
    const float sc = 0.08838834764831845f;
#pragma unroll
    for (int c = 0; c < 4; ++c) {
      const float* p0 = qp + c * 32 + hi * 8;
      float4 f0 = *(const float4*)(p0);
      float4 f1 = *(const float4*)(p0 + 4);
      float4 f2 = *(const float4*)(p0 + 16);
      float4 f3 = *(const float4*)(p0 + 20);
      qf[c].u[0] = pack2(f0.x * sc, f0.y * sc);
      qf[c].u[1] = pack2(f0.z * sc, f0.w * sc);
      qf[c].u[2] = pack2(f1.x * sc, f1.y * sc);
      qf[c].u[3] = pack2(f1.z * sc, f1.w * sc);
      qf[c].u[4] = pack2(f2.x * sc, f2.y * sc);
      qf[c].u[5] = pack2(f2.z * sc, f2.w * sc);
      qf[c].u[6] = pack2(f3.x * sc, f3.y * sc);
      qf[c].u[7] = pack2(f3.z * sc, f3.w * sc);
    }
  }

  v8f acco[8] = {};
  float mstat[8], lstat[8], alpha[8];
#pragma unroll
  for (int r = 0; r < 8; ++r) { mstat[r] = -__builtin_inff(); lstat[r] = 0.f; }

  for (int kb = 0; kb <= qb; kb += 64) {
    { // stage K, V tiles via async copies (bf16, pre-laid-out, no transposition)
      const unsigned short* kg = kbf + ((size_t)bh * S_ + kb) * HD_;
#pragma unroll
      for (int j = 0; j < 8; ++j) {
        int chunk = tid + j * 128;               // 64 rows x 16 chunks
        int row = chunk >> 4, c = chunk & 15;
        async_g2l_b128(lk0 + (unsigned)(row * 136 + c * 8) * 2, kg + row * HD_ + c * 8);
      }
      const unsigned short* vg = vbt + (size_t)bh * HD_ * S_ + kb;
#pragma unroll
      for (int j = 0; j < 8; ++j) {
        int chunk = tid + j * 128;               // 128 rows x 8 chunks
        int row = chunk >> 3, c = chunk & 7;
        async_g2l_b128(lv0 + (unsigned)(row * 72 + c * 8) * 2, vg + (size_t)row * S_ + c * 8);
      }
    }
    wait_async0();
    __syncthreads();

    // scores: 16 q-rows x 64 keys
    v8f accs[4] = {};
#pragma unroll
    for (int c = 0; c < 4; ++c)
#pragma unroll
      for (int ni = 0; ni < 4; ++ni) {
        Frag kf;
        int base = (ni * 16 + ln) * 136 + c * 32 + hi * 8;
        kf.q[0] = *(const uint4*)(lds_k + base);
        kf.q[1] = *(const uint4*)(lds_k + base + 16);
        accs[ni] = wmma_bf16(qf[c], kf, accs[ni]);
      }

    const int qmin = qb + w * 16;
    if (kb + 63 > qmin) {  // causal mask (diagonal tiles only)
#pragma unroll
      for (int ni = 0; ni < 4; ++ni)
#pragma unroll
        for (int r = 0; r < 8; ++r) {
          int key = kb + ni * 16 + ln;
          int qr = qmin + r + hi * 8;
          if (key > qr) accs[ni][r] = -1e30f;
        }
    }

    // online softmax (rows live in 16-lane halves; reduce with shfl_xor)
#pragma unroll
    for (int r = 0; r < 8; ++r) {
      float mx = mstat[r];
#pragma unroll
      for (int ni = 0; ni < 4; ++ni) mx = fmaxf(mx, accs[ni][r]);
      mx = fmaxf(mx, __shfl_xor(mx, 8, 32));
      mx = fmaxf(mx, __shfl_xor(mx, 4, 32));
      mx = fmaxf(mx, __shfl_xor(mx, 2, 32));
      mx = fmaxf(mx, __shfl_xor(mx, 1, 32));
      alpha[r] = exp2f((mstat[r] - mx) * 1.44269504f);
      mstat[r] = mx;
    }
#pragma unroll
    for (int r = 0; r < 8; ++r) {
      float s = 0.f;
#pragma unroll
      for (int ni = 0; ni < 4; ++ni) {
        float p = exp2f((accs[ni][r] - mstat[r]) * 1.44269504f);
        accs[ni][r] = p; s += p;
      }
      s += __shfl_xor(s, 8, 32);
      s += __shfl_xor(s, 4, 32);
      s += __shfl_xor(s, 2, 32);
      s += __shfl_xor(s, 1, 32);
      lstat[r] = lstat[r] * alpha[r] + s;
    }
#pragma unroll
    for (int nt = 0; nt < 8; ++nt)
#pragma unroll
      for (int r = 0; r < 8; ++r) acco[nt][r] *= alpha[r];

    // P (C layout) -> per-wave LDS -> A-layout fragments
#pragma unroll
    for (int ni = 0; ni < 4; ++ni)
#pragma unroll
      for (int r = 0; r < 8; ++r)
        lds_p[(w * 16 + r + hi * 8) * 72 + ni * 16 + ln] = f2bf(accs[ni][r]);

#pragma unroll
    for (int c2 = 0; c2 < 2; ++c2) {
      Frag pf;
      int pbase = (w * 16 + ln) * 72 + c2 * 32 + hi * 8;
      pf.q[0] = *(const uint4*)(lds_p + pbase);
      pf.q[1] = *(const uint4*)(lds_p + pbase + 16);
#pragma unroll
      for (int nt = 0; nt < 8; ++nt) {
        Frag vf;
        int vbase = (nt * 16 + ln) * 72 + c2 * 32 + hi * 8;
        vf.q[0] = *(const uint4*)(lds_v + vbase);
        vf.q[1] = *(const uint4*)(lds_v + vbase + 16);
        acco[nt] = wmma_bf16(pf, vf, acco[nt]);
      }
    }
    __syncthreads();
  }

#pragma unroll
  for (int r = 0; r < 8; ++r) lstat[r] = 1.0f / lstat[r];
#pragma unroll
  for (int nt = 0; nt < 8; ++nt)
#pragma unroll
    for (int r = 0; r < 8; ++r) {
      int qr = qb + w * 16 + r + hi * 8;
      attn[(size_t)(b * S_ + qr) * D_ + h * HD_ + nt * 16 + ln] =
          f2bf(acco[nt][r] * lstat[r]);
    }
}

// ------------------------------------------------ host orchestration
extern "C" void kernel_launch(void* const* d_in, const int* in_sizes, int n_in,
                              void* d_out, int out_size, void* d_ws, size_t ws_size,
                              hipStream_t stream) {
  const int*   positions = (const int*)d_in[0];
  const float* hidden    = (const float*)d_in[1];
  const float* ln1       = (const float*)d_in[2];
  const float* ln2       = (const float*)d_in[3];
  const float* Wqkv      = (const float*)d_in[4];
  const float* bqkv      = (const float*)d_in[5];
  const float* Wo        = (const float*)d_in[6];
  const float* W1        = (const float*)d_in[7];
  const float* W2        = (const float*)d_in[8];
  const float* Wout      = (const float*)d_in[9];
  float* out = (float*)d_out;

  char* p = (char*)d_ws;
  auto alloc = [&](size_t bytes) -> void* {
    void* r = (void*)p;
    p += (bytes + 255) & ~(size_t)255;
    return r;
  };
  const size_t NWQKV = (size_t)D_ * QKVN_;
  const size_t NWO   = (size_t)D_ * D_;
  const size_t NW12  = (size_t)D_ * FF_;
  const size_t NWOUT = (size_t)FF_ * D_;

  unsigned short* wqkv_t = (unsigned short*)alloc(NWQKV * 2);   // [3D][D]
  unsigned short* wo_t   = (unsigned short*)alloc(NWO * 2);     // [D][D]
  unsigned short* w1_t   = (unsigned short*)alloc(NW12 * 2);    // [FF][D]
  unsigned short* w2_t   = (unsigned short*)alloc(NW12 * 2);    // [FF][D]
  unsigned short* wout_t = (unsigned short*)alloc(NWOUT * 2);   // [D][FF]
  unsigned short* xn     = (unsigned short*)alloc((size_t)TOK_ * D_ * 2);
  float*          qkv    = (float*)alloc((size_t)TOK_ * QKVN_ * 4);
  unsigned short* kbf    = (unsigned short*)alloc((size_t)TOK_ * D_ * 2);  // [bh,s,hd]
  unsigned short* vbt    = (unsigned short*)alloc((size_t)TOK_ * D_ * 2);  // [bh,hd,s]
  unsigned short* attn   = (unsigned short*)alloc((size_t)TOK_ * D_ * 2);
  float*          hbuf   = (float*)alloc((size_t)TOK_ * D_ * 4);
  unsigned short* hn     = (unsigned short*)alloc((size_t)TOK_ * D_ * 2);
  unsigned short* act    = (unsigned short*)alloc((size_t)TOK_ * FF_ * 2);

  // weights: fp32 [K,N] -> bf16 transposed [N,K] (once per call)
  cvt_transpose<<<dim3(QKVN_ / 32, D_ / 32), 256, 0, stream>>>(Wqkv, wqkv_t, D_, QKVN_);
  cvt_transpose<<<dim3(D_ / 32, D_ / 32),    256, 0, stream>>>(Wo,   wo_t,   D_, D_);
  cvt_transpose<<<dim3(FF_ / 32, D_ / 32),   256, 0, stream>>>(W1,   w1_t,   D_, FF_);
  cvt_transpose<<<dim3(FF_ / 32, D_ / 32),   256, 0, stream>>>(W2,   w2_t,   D_, FF_);
  cvt_transpose<<<dim3(D_ / 32, FF_ / 32),   256, 0, stream>>>(Wout, wout_t, FF_, D_);

  rmsnorm_kernel<<<TOK_, 256, 0, stream>>>(hidden, ln1, xn);

  gemm_bf16<0><<<dim3(QKVN_ / 128, TOK_ / 64), 256, 0, stream>>>(
      xn, wqkv_t, bqkv, qkv, TOK_, QKVN_, D_);

  rope_kernel<<<(TOK_ * H_ * (HD_ / 2) + 255) / 256, 256, 0, stream>>>(positions, qkv);

  cvt_khead<<<(int)(((size_t)TOK_ * D_ / 4 + 255) / 256), 256, 0, stream>>>(qkv, kbf);
  vtrans<<<dim3(S_ / 32, HD_ / 32, B_ * H_), 256, 0, stream>>>(qkv, vbt);

  attn_kernel<<<B_ * H_ * (S_ / 64), 128, 0, stream>>>(qkv, kbf, vbt, attn);

  gemm_bf16<1><<<dim3(D_ / 128, TOK_ / 64), 256, 0, stream>>>(
      attn, wo_t, hidden, hbuf, TOK_, D_, D_);

  rmsnorm_kernel<<<TOK_, 256, 0, stream>>>(hbuf, ln2, hn);

  gemm_swiglu<<<dim3(FF_ / 128, TOK_ / 64), 256, 0, stream>>>(
      hn, w1_t, w2_t, act, TOK_, FF_, D_);

  gemm_bf16<1><<<dim3(D_ / 128, TOK_ / 64), 256, 0, stream>>>(
      act, wout_t, hbuf, out, TOK_, D_, FF_);
}
